// NPD_18253611008527
// MI455X (gfx1250) — matrix operand order repaired and checked by reading, not compiled, not run on hardware
//
#include <hip/hip_runtime.h>
#include <stdint.h>

// ---------- types ----------
typedef __attribute__((ext_vector_type(16))) __bf16 v16bf;
typedef __attribute__((ext_vector_type(8)))  float  v8f;
typedef __attribute__((ext_vector_type(4)))  unsigned int u32x4;
typedef __attribute__((ext_vector_type(8)))  int    i32x8;
typedef __attribute__((ext_vector_type(4)))  int    i32x4;

// ---------- problem constants ----------
#define FEAT    128
#define HIDDEN  256
#define PDIM    32
#define BATCH   4096
#define COLS    51
#define ROWS_PAD 64
#define EPSV    1e-5f

// padded row strides (break 64-bank LDS conflicts: +16B per row -> bank rot 4)
#define XS 136    // stride (ushorts) for x and W0 rows (128 + 8)
#define HS 264    // stride (ushorts) for h1/h2/W1/WU rows (256 + 8)
#define ES 33     // stride (floats) for e rows (32 + 1)

// ushort element offsets inside ws / LDS weight region (padded layout)
#define W0_OFF  0
#define W1_OFF  (HIDDEN*XS)                 // 34816
#define WU_OFF  (W1_OFF + HIDDEN*HS)        // 102400
#define W_TOTAL (WU_OFF + PDIM*HS)          // 110848 ushorts = 221696 B

// byte offsets inside dynamic LDS
#define LDS_W_BYTES (W_TOTAL*2)                  // 221696
#define LDS_X_OFF   LDS_W_BYTES
#define LDS_X_BYTES (ROWS_PAD*XS*2)              // 17408
#define LDS_H1_OFF  (LDS_X_OFF + LDS_X_BYTES)    // 239104
#define LDS_H_BYTES (ROWS_PAD*HS*2)              // 33792
#define LDS_H2_OFF  (LDS_H1_OFF + LDS_H_BYTES)   // 272896
#define LDS_E_OFF   (LDS_H2_OFF + LDS_H_BYTES)   // 306688
#define LDS_E_BYTES (ROWS_PAD*ES*4)              // 8448
#define LDS_TOTAL   (LDS_E_OFF + LDS_E_BYTES)    // 315136 B < 320KB

// ---------- helpers ----------
__device__ __forceinline__ unsigned short f2bf(float f) {
    unsigned u = __float_as_uint(f);
    unsigned r = u + 0x7FFFu + ((u >> 16) & 1u);   // round-to-nearest-even
    return (unsigned short)(r >> 16);
}

// tanh via hardware exp2/rcp: tanh(x) = 1 - 2/(exp2(2*log2e*x)+1)
__device__ __forceinline__ float fast_tanh(float x) {
    const float xc = fminf(fmaxf(x, -15.f), 15.f);
    const float e  = __builtin_amdgcn_exp2f(xc * 2.88539008177793f);
    return 1.f - 2.f * __builtin_amdgcn_rcpf(e + 1.f);
}

union Frag { uint4 q[2]; v16bf v; };

// A-matrix 16x32 bf16 fragment (lanes 0-15: K=k0..k0+7,k0+16..k0+23 of row;
// lanes 16-31: +8 shifted), from LDS row-major ushort array.
__device__ __forceinline__ v16bf load_afrag(const unsigned short* base, int stride,
                                            int row, int k0, int lane) {
    const int half = (lane >= 16) ? 8 : 0;
    const unsigned short* r = base + row * stride + k0 + half;
    Frag f;
    f.q[0] = *(const uint4*)(r);
    f.q[1] = *(const uint4*)(r + 16);
    return f.v;
}

// B-matrix 32x16 bf16 fragment: lane n (0-15) holds column n, K=k0..k0+7,k0+16..k0+23;
// lanes 16-31: +8 shifted.  B column n == weight row n (NT GEMM).
__device__ __forceinline__ v16bf load_bfrag(const unsigned short* base, int stride,
                                            int nrow, int k0, int lane) {
    const unsigned short* r = base + nrow * stride + k0 + ((lane >= 16) ? 8 : 0);
    Frag f;
    f.q[0] = *(const uint4*)(r);
    f.q[1] = *(const uint4*)(r + 16);
    return f.v;
}

// TDM: 1D tile copy (nq8 x 8-byte elements) from global bf16 weights to LDS.
__device__ __forceinline__ void tdm_load_weights(const void* gaddr, unsigned lds_off,
                                                 unsigned nq8) {
    unsigned long long ga = (unsigned long long)(uintptr_t)gaddr;
    u32x4 g0;
    g0[0] = 1u;                                             // count=1, user mode
    g0[1] = lds_off;                                        // lds_addr
    g0[2] = (unsigned)(ga & 0xFFFFFFFFu);                   // global_addr[31:0]
    g0[3] = (unsigned)((ga >> 32) & 0x1FFFFFFu) | 0x80000000u; // addr[56:32] | type=2
    i32x8 g1;
    g1[0] = (int)(3u << 16);                 // data_size = 8B, no multicast
    g1[1] = (int)((nq8 & 0xFFFFu) << 16);    // tensor_dim0[15:0] @ bits 63:48
    g1[2] = (int)((nq8 >> 16) & 0xFFFFu) | (1 << 16); // dim0 hi16, tensor_dim1=1
    g1[3] = (int)((nq8 & 0xFFFFu) << 16);    // tile_dim0 @ bits 127:112
    g1[4] = 1;                               // tile_dim1=1, tile_dim2=0
    g1[5] = (int)nq8;                        // tensor_dim0_stride[31:0]
    g1[6] = 0;
    g1[7] = 0;
    i32x4 z4 = {0, 0, 0, 0};
    i32x8 z8 = {0, 0, 0, 0, 0, 0, 0, 0};
    __builtin_amdgcn_tensor_load_to_lds(g0, g1, z4, z4, z8, 0);
}

#define WMMA_BF16(A, B, C) \
    __builtin_amdgcn_wmma_f32_16x16x32_bf16(false, (A), false, (B), (short)0, (C), false, false)

// ---------- prep: f32 weights -> padded bf16 workspace ----------
extern "C" __global__ __launch_bounds__(256)
void npd_prep(const float* __restrict__ W0, const float* __restrict__ W1,
              const float* __restrict__ WU, unsigned short* __restrict__ ws) {
    int i = blockIdx.x * blockDim.x + threadIdx.x;
    if (i >= W_TOTAL) return;
    float f = 0.f;
    if (i < W1_OFF) {
        const int row = i / XS, c = i % XS;
        if (c < FEAT) f = W0[row * FEAT + c];
    } else if (i < WU_OFF) {
        const int j = i - W1_OFF;
        const int row = j / HS, c = j % HS;
        if (c < HIDDEN) f = W1[row * HIDDEN + c];
    } else {
        const int j = i - WU_OFF;
        const int row = j / HS, c = j % HS;
        if (c < HIDDEN) f = WU[row * HIDDEN + c];
    }
    ws[i] = f2bf(f);
}

// ---------- main kernel: one block per batch row ----------
extern "C" __global__ __launch_bounds__(256)
void npd_main(const int* __restrict__ inputs, const float* __restrict__ dv,
              const float* __restrict__ b1, const float* __restrict__ bU,
              const unsigned short* __restrict__ wbf, float* __restrict__ out) {
    extern __shared__ unsigned char smem[];
    unsigned short* lw  = (unsigned short*)smem;
    unsigned short* lx  = (unsigned short*)(smem + LDS_X_OFF);
    unsigned short* lh1 = (unsigned short*)(smem + LDS_H1_OFF);
    unsigned short* lh2 = (unsigned short*)(smem + LDS_H2_OFF);
    float*          le  = (float*)(smem + LDS_E_OFF);

    const int tid  = threadIdx.x;
    const int lane = tid & 31;
    const int w    = tid >> 5;
    const int b    = blockIdx.x;
    const int l15  = lane & 15;
    const int hi8  = (lane >= 16) ? 8 : 0;

    // ---- stage 0: TDM DMA of padded bf16 weights into LDS (one wave issues) ----
    if (w == 0) {
        unsigned lds_base = (unsigned)(unsigned long long)(uintptr_t)smem;
        tdm_load_weights(wbf, lds_base, LDS_W_BYTES / 8);
    }

    // ---- stage 1: gather x rows, convert to bf16 in LDS (overlaps TDM) ----
    for (int i = tid; i < ROWS_PAD * (FEAT / 4); i += 256) {
        const int row = i >> 5;            // FEAT/4 == 32 chunks per row
        const int col = (i & 31) << 2;
        unsigned short v0 = 0, v1 = 0, v2 = 0, v3 = 0;
        if (row < COLS) {
            const int node = inputs[b * COLS + row];
            const float4 f = *(const float4*)(dv + (long long)node * FEAT + col);
            v0 = f2bf(f.x); v1 = f2bf(f.y); v2 = f2bf(f.z); v3 = f2bf(f.w);
        }
        uint2 pk;
        pk.x = (unsigned)v0 | ((unsigned)v1 << 16);
        pk.y = (unsigned)v2 | ((unsigned)v3 << 16);
        *(uint2*)(lx + row * XS + col) = pk;
    }

    if (w == 0) __builtin_amdgcn_s_wait_tensorcnt(0);
    __syncthreads();

    // ---- GEMM1: h1 = tanh(x @ W0^T)  M=64 N=256 K=128 ----
    {
        const int mt = (w & 3) * 16;
        const int ng = (w >> 2) * 8;   // 8 n-tiles per wave
        v8f acc[8] = {};
        #pragma unroll
        for (int kk = 0; kk < FEAT; kk += 32) {
            const v16bf a = load_afrag(lx, XS, mt + l15, kk, lane);
            #pragma unroll
            for (int t = 0; t < 8; ++t) {
                const v16bf bf = load_bfrag(lw + W0_OFF, XS, (ng + t) * 16 + l15, kk, lane);
                acc[t] = WMMA_BF16(a, bf, acc[t]);
            }
        }
        #pragma unroll
        for (int t = 0; t < 8; ++t) {
            const int colb = (ng + t) * 16 + l15;
            #pragma unroll
            for (int v = 0; v < 8; ++v)
                lh1[(mt + v + hi8) * HS + colb] = f2bf(fast_tanh(acc[t][v]));
        }
    }
    __syncthreads();

    // ---- GEMM2: h2 = tanh(h1 @ W1^T + b1)  M=64 N=256 K=256 ----
    {
        const int mt = (w & 3) * 16;
        const int ng = (w >> 2) * 8;
        v8f acc[8] = {};
        #pragma unroll
        for (int kk = 0; kk < HIDDEN; kk += 32) {
            const v16bf a = load_afrag(lh1, HS, mt + l15, kk, lane);
            #pragma unroll
            for (int t = 0; t < 8; ++t) {
                const v16bf bf = load_bfrag(lw + W1_OFF, HS, (ng + t) * 16 + l15, kk, lane);
                acc[t] = WMMA_BF16(a, bf, acc[t]);
            }
        }
        #pragma unroll
        for (int t = 0; t < 8; ++t) {
            const int colb = (ng + t) * 16 + l15;
            const float bias = b1[colb];
            #pragma unroll
            for (int v = 0; v < 8; ++v)
                lh2[(mt + v + hi8) * HS + colb] = f2bf(fast_tanh(acc[t][v] + bias));
        }
    }
    __syncthreads();

    // ---- GEMM3: e = h2 @ WU^T + bU  M=64 N=32 K=256 (one tile per wave) ----
    {
        const int mt = (w & 3) * 16;
        const int nt = (w >> 2) & 1;
        v8f acc = {};
        #pragma unroll
        for (int kk = 0; kk < HIDDEN; kk += 32) {
            const v16bf a  = load_afrag(lh2, HS, mt + l15, kk, lane);
            const v16bf bf = load_bfrag(lw + WU_OFF, HS, nt * 16 + l15, kk, lane);
            acc = WMMA_BF16(a, bf, acc);
        }
        const int colb = nt * 16 + l15;
        const float bias = bU[colb];
        #pragma unroll
        for (int v = 0; v < 8; ++v)
            le[(mt + v + hi8) * ES + colb] = acc[v] + bias;
    }
    __syncthreads();

    // ---- epilogue A: clip each embedding to the unit ball ----
    if (tid < COLS) {
        float e[PDIM];
        float s = 0.f;
        #pragma unroll
        for (int k = 0; k < PDIM; ++k) { e[k] = le[tid * ES + k]; s += e[k] * e[k]; }
        const float n = sqrtf(s);
        const float f = (n >= 1.0f) ? (n / (1.0f - EPSV)) : 1.0f;
        const float inv = 1.0f / f;
        #pragma unroll
        for (int k = 0; k < PDIM; ++k) le[tid * ES + k] = e[k] * inv;
    }
    __syncthreads();

    // ---- epilogue B: hyperbolic distance col0 -> col j ----
    if (tid >= 1 && tid < COLS) {
        float squ = 0.f, sqv = 0.f, sqd = 0.f;
        #pragma unroll
        for (int k = 0; k < PDIM; ++k) {
            const float sv = le[k];
            const float ov = le[tid * ES + k];
            squ += sv * sv;
            sqv += ov * ov;
            const float d = sv - ov;
            sqd += d * d;
        }
        squ = fminf(fmaxf(squ, 0.f), 1.f - EPSV);
        sqv = fminf(fmaxf(sqv, 0.f), 1.f - EPSV);
        const float xx = sqd / ((1.f - squ) * (1.f - sqv)) * 2.f + 1.f;
        const float z  = sqrtf(xx * xx - 1.f);
        out[b * (COLS - 1) + (tid - 1)] = -logf(xx + z);
    }
}

// ---------- host entry ----------
extern "C" void kernel_launch(void* const* d_in, const int* in_sizes, int n_in,
                              void* d_out, int out_size, void* d_ws, size_t ws_size,
                              hipStream_t stream) {
    const int*   inputs = (const int*)  d_in[0];
    const float* dv     = (const float*)d_in[1];
    const float* W0     = (const float*)d_in[2];
    const float* W1     = (const float*)d_in[3];
    const float* b1     = (const float*)d_in[4];
    const float* WU     = (const float*)d_in[5];
    const float* bU     = (const float*)d_in[6];
    unsigned short* wbf = (unsigned short*)d_ws;     // 221696 B of scratch
    float* out = (float*)d_out;

    npd_prep<<<(W_TOTAL + 255) / 256, 256, 0, stream>>>(W0, W1, WU, wbf);
    npd_main<<<BATCH, 256, LDS_TOTAL, stream>>>(inputs, dv, b1, bU, wbf, out);
}
